// Attention_20237885899372
// MI455X (gfx1250) — matrix operand order repaired and checked by reading, not compiled, not run on hardware
//
#include <hip/hip_runtime.h>
#include <hip/hip_bf16.h>

// ---------------------------------------------------------------------------
// Problem constants (match the JAX reference)
// ---------------------------------------------------------------------------
constexpr int B  = 4;
constexpr int T  = 2048;
constexpr int C  = 1024;
constexpr int H  = 16;
constexpr int HD = C / H;          // 64
constexpr int M  = B * T;          // 8192 rows for all projections
constexpr float SCALE = 0.125f;    // 1/sqrt(64)

// ---------------------------------------------------------------------------
// bf16 / WMMA fragment types
// ---------------------------------------------------------------------------
typedef __bf16 bf;
typedef __attribute__((ext_vector_type(16))) bf    v16bf;
typedef __attribute__((ext_vector_type(8)))  bf    v8bf;
typedef __attribute__((ext_vector_type(8)))  float v8f;

struct frag_bits { uint4 lo; uint4 hi; };   // 32 bytes == one v16bf

static __device__ __forceinline__ v16bf make_frag(uint4 lo, uint4 hi) {
    frag_bits fb{lo, hi};
    return __builtin_bit_cast(v16bf, fb);   // pure register reinterpretation
}

static __device__ __forceinline__ bf f2bf(float f) {
    unsigned u = __builtin_bit_cast(unsigned, f);
    u += 0x7FFFu + ((u >> 16) & 1u);        // round-to-nearest-even
    unsigned short h = (unsigned short)(u >> 16);
    return __builtin_bit_cast(bf, h);
}

static __device__ __forceinline__ v8f wmma_bf16(v16bf a, v16bf b, v8f c) {
    // v_wmma_f32_16x16x32_bf16  (D = A[16x32] * B[32x16] + C)
    return __builtin_amdgcn_wmma_f32_16x16x32_bf16(
        false, a, false, b, (short)0, c, false, false);
}

// A-matrix fragment (16x32, M x K), row-major source, leading dim `ld`.
// lane(0-15)=M row; element e -> K = e + (e&8) + 8*half.
static __device__ __forceinline__ v16bf load_a_frag(const bf* base, int ld, int lane) {
    const int row  = lane & 15;
    const int half = lane >> 4;
    const bf* p = base + (size_t)row * ld + half * 8;
    return make_frag(*(const uint4*)(p), *(const uint4*)(p + 16));
}

// B-matrix fragment (32x16, K x N) from an N x K row-major source.
// lane(0-15)=N col; element e -> K = e + (e&8) + 8*half.
static __device__ __forceinline__ v16bf load_b_frag(const bf* base, int ld, int lane) {
    const int n    = lane & 15;
    const int half = lane >> 4;
    const bf* p = base + (size_t)n * ld + half * 8;
    return make_frag(*(const uint4*)(p), *(const uint4*)(p + 16));
}

// --- gfx1250 async copy: per-lane 16B global -> LDS, tracked by ASYNCcnt ----
static __device__ __forceinline__ unsigned lds_off(const void* p) {
    return (unsigned)(size_t)p;    // low 32 bits of flat addr = LDS offset
}
static __device__ __forceinline__ void async_load_b128(unsigned dst_lds, const bf* src) {
    asm volatile("global_load_async_to_lds_b128 %0, %1, off"
                 :: "v"(dst_lds), "v"(src) : "memory");
}
static __device__ __forceinline__ void wait_asynccnt0() {
    asm volatile("s_wait_asynccnt 0x0" ::: "memory");
}

// ---------------------------------------------------------------------------
// Kernel 1: fp32 -> bf16 cast (8 elements / thread-step, b128 traffic)
// ---------------------------------------------------------------------------
__global__ void cast_f32_to_bf16(const float* __restrict__ in,
                                 bf* __restrict__ out, size_t n8) {
    size_t i = (size_t)blockIdx.x * blockDim.x + threadIdx.x;
    size_t stride = (size_t)gridDim.x * blockDim.x;
    const float4* f4 = (const float4*)in;
    for (; i < n8; i += stride) {
        float4 a = f4[2 * i];
        float4 b = f4[2 * i + 1];
        v8bf o;
        o[0] = f2bf(a.x); o[1] = f2bf(a.y); o[2] = f2bf(a.z); o[3] = f2bf(a.w);
        o[4] = f2bf(b.x); o[5] = f2bf(b.y); o[6] = f2bf(b.z); o[7] = f2bf(b.w);
        *(v8bf*)(out + 8 * i) = o;
    }
}

// ---------------------------------------------------------------------------
// Kernel 2: WMMA GEMM  out[M,N] = A[M,K] @ W[N,K]^T (+ bias)
// Block = 8 waves sharing one 64-row A stripe. The A k-slice (64x32 bf16,
// 4 KB) is staged to LDS with global_load_async_to_lds_b128 (one async op per
// wave = 512 B), double-buffered, one barrier per k-step. Each wave computes
// a 64x64 output tile: 16 WMMA per k-step.
// OUT_MODE: 0 = fp32 row-major, 1 = bf16 row-major,
//           2 = bf16 transposed to [B,H,HD,T] (for V)
// ---------------------------------------------------------------------------
template <int OUT_MODE>
__global__ __launch_bounds__(256)
void gemm_bias_wmma(const bf* __restrict__ A, const bf* __restrict__ W,
                    const float* __restrict__ bias, void* __restrict__ outp,
                    int Mdim, int Ndim, int Kdim) {
    __shared__ bf lds_a[2][64 * 32];        // 4 KB x 2 (double buffer)

    const int tid  = threadIdx.x;
    const int lane = tid & 31;
    const int wid  = blockIdx.x * (blockDim.x >> 5) + (tid >> 5);
    const int Nt   = Ndim >> 6;
    const int mt   = wid / Nt;              // identical for all 8 waves/block
    const int nt   = wid % Nt;
    const int mbase = mt * 64;
    const int nbase = nt * 64;

    // staging role of this thread: 16 bytes of the 64x32 A tile
    const int srow = tid >> 2;              // 0..63
    const int sseg = tid & 3;               // 0..3 (8 bf16 each)
    const unsigned dst0 = lds_off(&lds_a[0][srow * 32 + sseg * 8]);
    const unsigned dst1 = lds_off(&lds_a[1][srow * 32 + sseg * 8]);
    const bf* srcA = A + (size_t)(mbase + srow) * Kdim + sseg * 8;

    v8f zero = {};
    v8f acc[4][4];
#pragma unroll
    for (int i = 0; i < 4; ++i)
#pragma unroll
        for (int j = 0; j < 4; ++j) acc[i][j] = zero;

    for (int kb = 0; kb < Kdim; kb += 32) {
        const int p = (kb >> 5) & 1;
        // --- async stage A k-slice into LDS (ASYNCcnt-tracked) ---
        async_load_b128(p ? dst1 : dst0, srcA + kb);
        wait_asynccnt0();
        __syncthreads();

        // prefetch next W stripe toward the caches (global_prefetch_b8)
        if (kb + 32 < Kdim)
            __builtin_prefetch(W + (size_t)(nbase + (lane & 3) * 16 + (lane >> 2)) * Kdim + kb + 32, 0, 1);

        v16bf bfr[4];
#pragma unroll
        for (int j = 0; j < 4; ++j)
            bfr[j] = load_b_frag(W + (size_t)(nbase + j * 16) * Kdim + kb, Kdim, lane);

        const bf* atile = lds_a[p];
#pragma unroll
        for (int i = 0; i < 4; ++i) {
            v16bf afr = load_a_frag(atile + i * 16 * 32, 32, lane);  // ds_load_b128 x2
#pragma unroll
            for (int j = 0; j < 4; ++j)
                acc[i][j] = wmma_bf16(afr, bfr[j], acc[i][j]);
        }
    }

    const int half = lane >> 4;
    const int ncol = lane & 15;
#pragma unroll
    for (int j = 0; j < 4; ++j) {
        const int n = nbase + j * 16 + ncol;
        const float bv = bias ? bias[n] : 0.0f;
        if constexpr (OUT_MODE == 2) {
            // V path: write V^T laid out [B, H, HD, T]; 8 accumulator rows of
            // one (i,j) are 8 consecutive t -> one 16-byte store.
            const int h  = n / HD;
            const int hd = n % HD;
#pragma unroll
            for (int i = 0; i < 4; ++i) {
                const int m0 = mbase + i * 16 + half * 8;   // tile never crosses b
                const int b_ = m0 / T;
                const int t0 = m0 % T;
                v8bf o;
#pragma unroll
                for (int e = 0; e < 8; ++e) o[e] = f2bf(acc[i][j][e] + bv);
                *(v8bf*)((bf*)outp + ((size_t)((b_ * H + h) * HD + hd)) * T + t0) = o;
            }
        } else {
#pragma unroll
            for (int i = 0; i < 4; ++i) {
#pragma unroll
                for (int e = 0; e < 8; ++e) {
                    const int m = mbase + i * 16 + half * 8 + e;
                    const float v = acc[i][j][e] + bv;
                    if constexpr (OUT_MODE == 1)
                        ((bf*)outp)[(size_t)m * Ndim + n] = f2bf(v);
                    else
                        ((float*)outp)[(size_t)m * Ndim + n] = v;
                }
            }
        }
    }
}

// ---------------------------------------------------------------------------
// Kernel 3: causal flash attention. 8 independent waves per 256-thread block,
// each wave owns one (b, h, 16-row query tile) and a private 16x64 LDS P-tile.
// Keys streamed in chunks of 64: 8 score WMMAs + one softmax pass + 8 PV WMMAs
// per chunk. No block-wide sync.
// ---------------------------------------------------------------------------
__global__ __launch_bounds__(256)
void flash_attn_wmma(const bf* __restrict__ qb, const bf* __restrict__ kb_,
                     const bf* __restrict__ vT, bf* __restrict__ yb) {
    __shared__ bf ptile_all[8][16 * 64];    // 2 KB per wave

    const int lane = threadIdx.x & 31;
    const int wv   = threadIdx.x >> 5;
    bf* ptile = ptile_all[wv];

    const int half = lane >> 4;
    const int ncol = lane & 15;

    const int Tt = T / 16;
    const int id = blockIdx.x * 8 + wv;
    const int qt = id % Tt;
    const int h  = (id / Tt) % H;
    const int b  = id / (Tt * H);
    const int qbase = qt * 16;

    // Q fragments for this tile (registers for the whole loop)
    const bf* qptr = qb + ((size_t)(b * T + qbase)) * C + h * HD;
    const v16bf q0 = load_a_frag(qptr,      C, lane);   // k-dim 0..31
    const v16bf q1 = load_a_frag(qptr + 32, C, lane);   // k-dim 32..63

    v8f zero = {};
    v8f acc[4];
#pragma unroll
    for (int t = 0; t < 4; ++t) acc[t] = zero;
    float mrow[8], lrow[8];
#pragma unroll
    for (int e = 0; e < 8; ++e) { mrow[e] = -1e30f; lrow[e] = 0.0f; }

    for (int kc = 0; kc <= qbase + 15; kc += 64) {
        // ---- scores S[16 x 64] = Q @ K^T (four 16x16 n-tiles) ----
        v8f s[4];
#pragma unroll
        for (int j = 0; j < 4; ++j) {
            const bf* kptr = kb_ + ((size_t)(b * T + kc + j * 16)) * C + h * HD;
            v16bf bk0 = load_b_frag(kptr,      C, lane);   // d 0..31
            v16bf bk1 = load_b_frag(kptr + 32, C, lane);   // d 32..63
            v8f t = zero;
            t = wmma_bf16(q0, bk0, t);
            t = wmma_bf16(q1, bk1, t);
            s[j] = t;
        }

        const bool needmask = (kc + 63 > qbase);
#pragma unroll
        for (int j = 0; j < 4; ++j) {
#pragma unroll
            for (int e = 0; e < 8; ++e) {
                float sv = s[j][e] * SCALE;
                if (needmask) {
                    const int key  = kc + j * 16 + ncol;
                    const int qrow = qbase + half * 8 + e;
                    if (key > qrow) sv = -1e30f;
                }
                s[j][e] = sv;
            }
        }

        // ---- online softmax (row reductions stay inside 16-lane halves) ----
#pragma unroll
        for (int e = 0; e < 8; ++e) {
            float rm = fmaxf(fmaxf(s[0][e], s[1][e]), fmaxf(s[2][e], s[3][e]));
            rm = fmaxf(rm, __shfl_xor(rm, 1, 32));
            rm = fmaxf(rm, __shfl_xor(rm, 2, 32));
            rm = fmaxf(rm, __shfl_xor(rm, 4, 32));
            rm = fmaxf(rm, __shfl_xor(rm, 8, 32));
            const float mnew = fmaxf(mrow[e], rm);
            const float corr = __expf(mrow[e] - mnew);
            const float p0 = __expf(s[0][e] - mnew);
            const float p1 = __expf(s[1][e] - mnew);
            const float p2 = __expf(s[2][e] - mnew);
            const float p3 = __expf(s[3][e] - mnew);
            float rs = (p0 + p1) + (p2 + p3);
            rs += __shfl_xor(rs, 1, 32);
            rs += __shfl_xor(rs, 2, 32);
            rs += __shfl_xor(rs, 4, 32);
            rs += __shfl_xor(rs, 8, 32);
            lrow[e] = lrow[e] * corr + rs;
            mrow[e] = mnew;
#pragma unroll
            for (int t = 0; t < 4; ++t) acc[t][e] = acc[t][e] * corr;
            // spill P (C-layout) to LDS for the A-layout reload
            const int r = half * 8 + e;
            ptile[r * 64 + ncol]      = f2bf(p0);
            ptile[r * 64 + 16 + ncol] = f2bf(p1);
            ptile[r * 64 + 32 + ncol] = f2bf(p2);
            ptile[r * 64 + 48 + ncol] = f2bf(p3);
        }

        // ---- reload P as A fragments (same-wave LDS ops are in-order) ----
        const v16bf pf0 = load_a_frag(ptile,      64, lane);  // keys kc..kc+31
        const v16bf pf1 = load_a_frag(ptile + 32, 64, lane);  // keys kc+32..kc+63

        // ---- O += P @ V : V^T is [B,H,HD,T], B-frags contiguous ----
#pragma unroll
        for (int t = 0; t < 4; ++t) {
            const bf* vptr = vT + ((size_t)((b * H + h) * HD + t * 16)) * T + kc;
            v16bf bv0 = load_b_frag(vptr,      T, lane);
            v16bf bv1 = load_b_frag(vptr + 32, T, lane);
            acc[t] = wmma_bf16(pf0, bv0, acc[t]);
            acc[t] = wmma_bf16(pf1, bv1, acc[t]);
        }
    }

    // ---- epilogue: normalize and write y tile in bf16 ----
#pragma unroll
    for (int t = 0; t < 4; ++t) {
#pragma unroll
        for (int e = 0; e < 8; ++e) {
            const float o = acc[t][e] / lrow[e];
            const int m = qbase + half * 8 + e;
            yb[((size_t)(b * T + m)) * C + h * HD + t * 16 + ncol] = f2bf(o);
        }
    }
}

// ---------------------------------------------------------------------------
// Host-side launcher
// ---------------------------------------------------------------------------
extern "C" void kernel_launch(void* const* d_in, const int* in_sizes, int n_in,
                              void* d_out, int out_size, void* d_ws, size_t ws_size,
                              hipStream_t stream) {
    const float* x  = (const float*)d_in[0];
    const float* Wq = (const float*)d_in[1];
    const float* bq = (const float*)d_in[2];
    const float* Wk = (const float*)d_in[3];
    const float* bk = (const float*)d_in[4];
    const float* Wv = (const float*)d_in[5];
    const float* bv = (const float*)d_in[6];
    const float* Wp = (const float*)d_in[7];
    float* out = (float*)d_out;

    // workspace layout (bf16 buffers)
    char* w = (char*)d_ws;
    bf* xb  = (bf*)w; w += (size_t)M * C * 2;           // 16 MB
    bf* wqb = (bf*)w; w += (size_t)C * C * 2;           //  2 MB
    bf* wkb = (bf*)w; w += (size_t)C * C * 2;
    bf* wvb = (bf*)w; w += (size_t)C * C * 2;
    bf* wpb = (bf*)w; w += (size_t)C * C * 2;
    bf* qb  = (bf*)w; w += (size_t)M * C * 2;           // 16 MB
    bf* kbm = (bf*)w; w += (size_t)M * C * 2;           // 16 MB
    bf* vT  = (bf*)w; w += (size_t)M * C * 2;           // 16 MB, [B,H,HD,T]
    bf* yb  = (bf*)w; w += (size_t)M * C * 2;           // 16 MB

    // 1) casts (8 elements per thread-step)
    cast_f32_to_bf16<<<1024, 256, 0, stream>>>(x,  xb,  (size_t)M * C / 8);
    cast_f32_to_bf16<<<512,  256, 0, stream>>>(Wq, wqb, (size_t)C * C / 8);
    cast_f32_to_bf16<<<512,  256, 0, stream>>>(Wk, wkb, (size_t)C * C / 8);
    cast_f32_to_bf16<<<512,  256, 0, stream>>>(Wv, wvb, (size_t)C * C / 8);
    cast_f32_to_bf16<<<512,  256, 0, stream>>>(Wp, wpb, (size_t)C * C / 8);

    // 2) Q/K/V projections: (M/64)*(C/64) waves, 8 waves per block
    const int gemm_blocks = (M / 64) * (C / 64) / 8;   // 256
    gemm_bias_wmma<1><<<gemm_blocks, 256, 0, stream>>>(xb, wqb, bq, qb,  M, C, C);
    gemm_bias_wmma<1><<<gemm_blocks, 256, 0, stream>>>(xb, wkb, bk, kbm, M, C, C);
    gemm_bias_wmma<2><<<gemm_blocks, 256, 0, stream>>>(xb, wvb, bv, vT,  M, C, C);

    // 3) causal flash attention: 8 waves per block, one (b,h,qtile) per wave
    flash_attn_wmma<<<B * H * (T / 16) / 8, 256, 0, stream>>>(qb, kbm, vT, yb);

    // 4) output projection (fp32 result, no bias)
    gemm_bias_wmma<0><<<gemm_blocks, 256, 0, stream>>>(yb, wpb, nullptr, out, M, C, C);
}
